// GraphTransformerSelfAttLayer_34926674051620
// MI455X (gfx1250) — matrix operand order repaired and checked by reading, not compiled, not run on hardware
//
#include <hip/hip_runtime.h>

// ---------------------------------------------------------------------------
// GraphTransformer self-attention layer for gfx1250 (MI455X).
// All GEMMs via v_wmma_f32_16x16x32_f16 (wave32, 16x16 tiles, f32 accum).
//
// Fragment layouts used here (wave32; k order is this code's choice, not the ISA table order):
//  A (16x32 f16): lane L -> row m=L&15, half hg=L>>4.
//    halves [0..15] = K(kk + hg*16 + i)   (contiguous, same as B)
//  B (32x16 f16): lane L -> col n=L&15, half hg=L>>4.
//    halves [0..15] = K(kk + hg*16 + i)   (contiguous -> weights stored
//    column-major [n][Kpad] so each lane does one 32-byte load)
//  C/D (16x16 f32, 8 VGPRs): c[r] at lane L = C[(L<16 ? r : r+8)][L&15]
// ---------------------------------------------------------------------------

typedef __attribute__((ext_vector_type(16))) _Float16 v16h;
typedef __attribute__((ext_vector_type(8)))  _Float16 v8h;
typedef __attribute__((ext_vector_type(8)))  float    v8f;

#define WPB 4          // waves per block (wave32)
#define TPB (WPB * 32)

static __device__ __forceinline__ v16h load_a_frag(const _Float16* A, int lda, int kk, int lane) {
  int m = lane & 15, hg = (lane >> 4) & 1;
  const _Float16* p0 = A + m * lda + kk + hg * 16;
  const _Float16* p1 = A + m * lda + kk + 8 + hg * 16;
  v8h lo = *(const v8h*)p0;
  v8h hi = *(const v8h*)p1;
  v16h r;
#pragma unroll
  for (int i = 0; i < 8; i++) { r[i] = lo[i]; r[i + 8] = hi[i]; }
  return r;
}

static __device__ __forceinline__ v16h load_b_frag(const _Float16* Wcm, int Kpad, int nn, int kk, int lane) {
  int n = nn + (lane & 15), hg = (lane >> 4) & 1;
  return *(const v16h*)(Wcm + (size_t)n * Kpad + kk + hg * 16);
}

template <int NT>
static __device__ __forceinline__ void wmma_gemm(const _Float16* A, int lda, int K,
                                                 const _Float16* Wcm, int Kpad,
                                                 v8f (&acc)[NT], int lane) {
  for (int kk = 0; kk < K; kk += 32) {
    v16h a = load_a_frag(A, lda, kk, lane);
#pragma unroll
    for (int nt = 0; nt < NT; nt++) {
      v16h b = load_b_frag(Wcm, Kpad, nt * 16, kk, lane);
      acc[nt] = __builtin_amdgcn_wmma_f32_16x16x32_f16(false, a, false, b,
                                                       (short)0, acc[nt], false, false);
    }
  }
}

// Dual-output GEMM: one A-fragment feeds both weight matrices (K & V MLP
// layer 1 share the same kv_input tile) -> halves LDS A-fragment traffic.
template <int NT>
static __device__ __forceinline__ void wmma_gemm_dual(const _Float16* A, int lda, int K,
                                                      const _Float16* W0cm, const _Float16* W1cm,
                                                      int Kpad,
                                                      v8f (&acc0)[NT], v8f (&acc1)[NT], int lane) {
  for (int kk = 0; kk < K; kk += 32) {
    v16h a = load_a_frag(A, lda, kk, lane);
#pragma unroll
    for (int nt = 0; nt < NT; nt++) {
      v16h b = load_b_frag(W0cm, Kpad, nt * 16, kk, lane);
      acc0[nt] = __builtin_amdgcn_wmma_f32_16x16x32_f16(false, a, false, b,
                                                        (short)0, acc0[nt], false, false);
    }
#pragma unroll
    for (int nt = 0; nt < NT; nt++) {
      v16h b = load_b_frag(W1cm, Kpad, nt * 16, kk, lane);
      acc1[nt] = __builtin_amdgcn_wmma_f32_16x16x32_f16(false, a, false, b,
                                                        (short)0, acc1[nt], false, false);
    }
  }
}

template <int NT>
static __device__ __forceinline__ void store_c_relu_f16(const v8f (&acc)[NT],
                                                        const float* __restrict__ bias,
                                                        _Float16* out, int ldo, int lane) {
  int n = lane & 15, hg = (lane >> 4) & 1;
#pragma unroll
  for (int nt = 0; nt < NT; nt++) {
    float bb = bias[nt * 16 + n];
#pragma unroll
    for (int r = 0; r < 8; r++) {
      int mm = hg ? (r + 8) : r;
      float x = acc[nt][r] + bb;
      x = fmaxf(x, 0.0f);
      out[mm * ldo + nt * 16 + n] = (_Float16)x;
    }
  }
}

static __device__ __forceinline__ void atomic_fadd(float* p, float v) {
  __hip_atomic_fetch_add(p, v, __ATOMIC_RELAXED, __HIP_MEMORY_SCOPE_AGENT);
}

// --- weight prep: fp32 row-major [K][N] -> f16 column-major [N][Kpad] -------
__global__ void prep_w_kernel(const float* __restrict__ src, _Float16* __restrict__ dst,
                              int K, int Nn, int Kpad) {
  int tid = blockIdx.x * blockDim.x + threadIdx.x;
  if (tid >= Nn * Kpad) return;
  int n = tid / Kpad, k = tid % Kpad;
  dst[tid] = (k < K) ? (_Float16)src[(size_t)k * Nn + n] : (_Float16)0.0f;
}

// --- q = MLP(h): [16,64]@[64,64] relu @[64,64] ------------------------------
__global__ __launch_bounds__(TPB) void q_mlp_kernel(
    const float* __restrict__ h,
    const _Float16* __restrict__ Wq1h, const float* __restrict__ bq1,
    const _Float16* __restrict__ Wq2h, const float* __restrict__ bq2,
    float* __restrict__ qbuf, int Ntiles) {
  __shared__ _Float16 s_a[WPB][16 * 64];
  __shared__ _Float16 s_h[WPB][16 * 64];
  int w = threadIdx.x >> 5, lane = threadIdx.x & 31;
  int tile = blockIdx.x * WPB + w;
  if (tile >= Ntiles) return;
  size_t n0 = (size_t)tile * 16;
  int m = lane & 15, hg = (lane >> 4) & 1;
  for (int c = hg * 32; c < hg * 32 + 32; c++)
    s_a[w][m * 64 + c] = (_Float16)h[(n0 + m) * 64 + c];
  v8f a1[4] = {};
  wmma_gemm<4>(s_a[w], 64, 64, Wq1h, 64, a1, lane);
  store_c_relu_f16<4>(a1, bq1, s_h[w], 64, lane);
  v8f a2[4] = {};
  wmma_gemm<4>(s_h[w], 64, 64, Wq2h, 64, a2, lane);
  int n = lane & 15;
#pragma unroll
  for (int nt = 0; nt < 4; nt++) {
    float bb = bq2[nt * 16 + n];
#pragma unroll
    for (int r = 0; r < 8; r++) {
      int mm = hg ? (r + 8) : r;
      qbuf[(n0 + mm) * 64 + nt * 16 + n] = a2[nt][r] + bb;
    }
  }
}

// --- per-edge: kv gather, K & V MLPs (fused layer-1), scores + seg-max ------
__global__ __launch_bounds__(TPB) void edge_kv_kernel(
    const float* __restrict__ h, const float* __restrict__ eattr,
    const long long* __restrict__ srcp, const long long* __restrict__ dstp,
    const _Float16* __restrict__ Wk1h, const float* __restrict__ bk1,
    const _Float16* __restrict__ Wk2h, const float* __restrict__ bk2,
    const _Float16* __restrict__ Wv1h, const float* __restrict__ bv1,
    const _Float16* __restrict__ Wv2h, const float* __restrict__ bv2,
    const float* __restrict__ qbuf,
    float* __restrict__ scores, _Float16* __restrict__ vbuf,
    unsigned* __restrict__ smax, int Etiles) {
  __shared__ _Float16 s_kv[WPB][16 * 160];   // kv_input padded 144 -> 160
  __shared__ _Float16 s_h2[WPB][16 * 64];    // hidden (reused for K then V)
  __shared__ float    s_kf[WPB][16 * 64];    // K tile (f32, for scores)
  __shared__ int      s_dst[WPB][16];
  int w = threadIdx.x >> 5, lane = threadIdx.x & 31;
  int tile = blockIdx.x * WPB + w;
  if (tile >= Etiles) return;
  size_t e0 = (size_t)tile * 16;
  int m = lane & 15, hg = (lane >> 4) & 1;
  size_t ee = e0 + m;
  if (hg == 0) {
    int di = (int)dstp[ee];
    s_dst[w][m] = di;
    for (int c = 0; c < 16; c++) s_kv[w][m * 160 + c]       = (_Float16)eattr[ee * 16 + c];
    for (int c = 0; c < 64; c++) s_kv[w][m * 160 + 16 + c]  = (_Float16)h[(size_t)di * 64 + c];
  } else {
    int si = (int)srcp[ee];
    for (int c = 0; c < 64; c++) s_kv[w][m * 160 + 80 + c]  = (_Float16)h[(size_t)si * 64 + c];
    for (int c = 0; c < 16; c++) s_kv[w][m * 160 + 144 + c] = (_Float16)0.0f;
  }
  // ---- fused layer-1: hidK = kv@Wk1, hidV = kv@Wv1 (shared A fragments) ----
  v8f ak1[4] = {}, av1[4] = {};
  wmma_gemm_dual<4>(s_kv[w], 160, 160, Wk1h, Wv1h, 160, ak1, av1, lane);
  // ---- K layer-2 ----
  store_c_relu_f16<4>(ak1, bk1, s_h2[w], 64, lane);
  v8f ak2[4] = {};
  wmma_gemm<4>(s_h2[w], 64, 64, Wk2h, 64, ak2, lane);
  {
    int n = lane & 15;
#pragma unroll
    for (int nt = 0; nt < 4; nt++) {
      float bb = bk2[nt * 16 + n];
#pragma unroll
      for (int r = 0; r < 8; r++) {
        int mm = hg ? (r + 8) : r;
        s_kf[w][mm * 64 + nt * 16 + n] = ak2[nt][r] + bb;
      }
    }
  }
  // ---- V layer-2 -> global f16 ----
  store_c_relu_f16<4>(av1, bv1, s_h2[w], 64, lane);
  v8f av2[4] = {};
  wmma_gemm<4>(s_h2[w], 64, 64, Wv2h, 64, av2, lane);
  {
    int n = lane & 15;
#pragma unroll
    for (int nt = 0; nt < 4; nt++) {
      float bb = bv2[nt * 16 + n];
#pragma unroll
      for (int r = 0; r < 8; r++) {
        int mm = hg ? (r + 8) : r;
        vbuf[(e0 + mm) * 64 + nt * 16 + n] = (_Float16)(av2[nt][r] + bb);
      }
    }
  }
  // ---- scores[e][head] = dot(q[dst], K) / sqrt(16); fused seg-max atomic ---
  int dn = s_dst[w][m];
#pragma unroll
  for (int t = 0; t < 2; t++) {
    int head = hg * 2 + t;
    float s = 0.0f;
#pragma unroll
    for (int j = 0; j < 16; j++)
      s += qbuf[(size_t)dn * 64 + head * 16 + j] * s_kf[w][m * 64 + head * 16 + j];
    s *= 0.25f;
    scores[(e0 + m) * 4 + head] = s;
    unsigned u = __float_as_uint(s);
    u = (u & 0x80000000u) ? ~u : (u | 0x80000000u);   // order-preserving encode
    atomicMax(&smax[dn * 4 + head], u);
  }
}

// --- fused exp + denominator + weighted-V accumulation (native f32 atomics) -
__global__ void scatter_kernel(const float* __restrict__ scores,
                               const long long* __restrict__ dstp,
                               const unsigned* __restrict__ smax,
                               const _Float16* __restrict__ vbuf,
                               float* __restrict__ denom, float* __restrict__ agg, int E4) {
  int tid = blockIdx.x * blockDim.x + threadIdx.x;
  if (tid >= E4) return;
  int eI = tid >> 2, hh = tid & 3;
  int dn = (int)dstp[eI];
  unsigned u = smax[dn * 4 + hh];
  float mx = (u & 0x80000000u) ? __uint_as_float(u ^ 0x80000000u) : __uint_as_float(~u);
  float ex = __expf(scores[tid] - mx);
  atomic_fadd(&denom[dn * 4 + hh], ex);
  v16h vv = *(const v16h*)(vbuf + (size_t)eI * 64 + hh * 16);   // 32B vector load
  float* ap = agg + (size_t)dn * 64 + hh * 16;
#pragma unroll
  for (int j = 0; j < 16; j++) atomic_fadd(&ap[j], ex * (float)vv[j]);
}

// --- x = h + (agg/denom)@WO + bO ; accumulate BN1 stats ---------------------
__global__ __launch_bounds__(TPB) void node_proj_kernel(
    const float* __restrict__ h, const float* __restrict__ agg,
    const float* __restrict__ denom,
    const _Float16* __restrict__ WOh, const float* __restrict__ bO,
    float* __restrict__ xpre, float* __restrict__ bnsum, float* __restrict__ bnsq, int Ntiles) {
  __shared__ _Float16 s_a[WPB][16 * 64];
  int w = threadIdx.x >> 5, lane = threadIdx.x & 31;
  int tile = blockIdx.x * WPB + w;
  if (tile >= Ntiles) return;
  size_t n0 = (size_t)tile * 16;
  int m = lane & 15, hg = (lane >> 4) & 1;
  for (int c = hg * 32; c < hg * 32 + 32; c++) {
    float dnm = denom[(n0 + m) * 4 + (c >> 4)];
    float hv = (dnm > 0.0f) ? agg[(n0 + m) * 64 + c] / dnm : 0.0f;
    s_a[w][m * 64 + c] = (_Float16)hv;
  }
  v8f acc[4] = {};
  wmma_gemm<4>(s_a[w], 64, 64, WOh, 64, acc, lane);
  int n = lane & 15;
#pragma unroll
  for (int nt = 0; nt < 4; nt++) {
    float bb = bO[nt * 16 + n];
    float ps = 0.0f, ss = 0.0f;
#pragma unroll
    for (int r = 0; r < 8; r++) {
      int mm = hg ? (r + 8) : r;
      float x = h[(n0 + mm) * 64 + nt * 16 + n] + acc[nt][r] + bb;
      xpre[(n0 + mm) * 64 + nt * 16 + n] = x;
      ps += x; ss += x * x;
    }
    atomic_fadd(&bnsum[nt * 16 + n], ps);
    atomic_fadd(&bnsq[nt * 16 + n], ss);
  }
}

// --- BN moments -> mean + rsqrt(var+eps) ------------------------------------
__global__ void bn_stats_kernel(const float* __restrict__ sum, const float* __restrict__ sq,
                                float* __restrict__ mu, float* __restrict__ rs, float invN) {
  int t = threadIdx.x;
  if (t >= 64) return;
  float mm = sum[t] * invN;
  float vv = sq[t] * invN - mm * mm;
  mu[t] = mm;
  rs[t] = rsqrtf(vv + 1e-5f);
}

// --- BN1 + FFN + residual ; accumulate BN2 stats ----------------------------
__global__ __launch_bounds__(TPB) void ffn_kernel(
    const float* __restrict__ xpre,
    const float* __restrict__ mu1, const float* __restrict__ rs1,
    const float* __restrict__ g1, const float* __restrict__ be1,
    const _Float16* __restrict__ W1h, const float* __restrict__ b1,
    const _Float16* __restrict__ W2h, const float* __restrict__ b2,
    float* __restrict__ ypre, float* __restrict__ bnsum2, float* __restrict__ bnsq2, int Ntiles) {
  __shared__ _Float16 s_x[WPB][16 * 64];
  __shared__ _Float16 s_hid[WPB][16 * 128];
  __shared__ float    s_xf[WPB][16 * 64];
  int w = threadIdx.x >> 5, lane = threadIdx.x & 31;
  int tile = blockIdx.x * WPB + w;
  if (tile >= Ntiles) return;
  size_t n0 = (size_t)tile * 16;
  int m = lane & 15, hg = (lane >> 4) & 1;
  for (int c = hg * 32; c < hg * 32 + 32; c++) {
    float xn = (xpre[(n0 + m) * 64 + c] - mu1[c]) * rs1[c] * g1[c] + be1[c];
    s_xf[w][m * 64 + c] = xn;
    s_x[w][m * 64 + c] = (_Float16)xn;
  }
  v8f a1[8] = {};
  wmma_gemm<8>(s_x[w], 64, 64, W1h, 64, a1, lane);
  store_c_relu_f16<8>(a1, b1, s_hid[w], 128, lane);
  v8f a2[4] = {};
  wmma_gemm<4>(s_hid[w], 128, 128, W2h, 128, a2, lane);
  int n = lane & 15;
#pragma unroll
  for (int nt = 0; nt < 4; nt++) {
    float bb = b2[nt * 16 + n];
    float ps = 0.0f, ss = 0.0f;
#pragma unroll
    for (int r = 0; r < 8; r++) {
      int mm = hg ? (r + 8) : r;
      float y = s_xf[w][mm * 64 + nt * 16 + n] + a2[nt][r] + bb;
      ypre[(n0 + mm) * 64 + nt * 16 + n] = y;
      ps += y; ss += y * y;
    }
    atomic_fadd(&bnsum2[nt * 16 + n], ps);
    atomic_fadd(&bnsq2[nt * 16 + n], ss);
  }
}

// --- final BN2 apply --------------------------------------------------------
__global__ void finalize_kernel(const float* __restrict__ ypre,
                                const float* __restrict__ mu, const float* __restrict__ rs,
                                const float* __restrict__ g, const float* __restrict__ b,
                                float* __restrict__ out, int total) {
  int tid = blockIdx.x * blockDim.x + threadIdx.x;
  if (tid >= total) return;
  int c = tid & 63;
  out[tid] = (ypre[tid] - mu[c]) * rs[c] * g[c] + b[c];
}

// ---------------------------------------------------------------------------
extern "C" void kernel_launch(void* const* d_in, const int* in_sizes, int n_in,
                              void* d_out, int out_size, void* d_ws, size_t ws_size,
                              hipStream_t stream) {
  const float*     h    = (const float*)d_in[0];
  const float*     ea   = (const float*)d_in[1];
  const long long* eidx = (const long long*)d_in[2];
  const float* Wk1 = (const float*)d_in[3];  const float* bk1 = (const float*)d_in[4];
  const float* Wk2 = (const float*)d_in[5];  const float* bk2 = (const float*)d_in[6];
  const float* Wv1 = (const float*)d_in[7];  const float* bv1 = (const float*)d_in[8];
  const float* Wv2 = (const float*)d_in[9];  const float* bv2 = (const float*)d_in[10];
  const float* Wq1 = (const float*)d_in[11]; const float* bq1 = (const float*)d_in[12];
  const float* Wq2 = (const float*)d_in[13]; const float* bq2 = (const float*)d_in[14];
  const float* WO  = (const float*)d_in[15]; const float* bO  = (const float*)d_in[16];
  const float* W1  = (const float*)d_in[17]; const float* b1  = (const float*)d_in[18];
  const float* W2  = (const float*)d_in[19]; const float* b2  = (const float*)d_in[20];
  const float* g1  = (const float*)d_in[21]; const float* be1 = (const float*)d_in[22];
  const float* g2  = (const float*)d_in[23]; const float* be2 = (const float*)d_in[24];
  float* out = (float*)d_out;

  const int N = in_sizes[0] / 64;        // 50000
  const int E = in_sizes[1] / 16;        // 800000
  const int Ntiles = N / 16, Etiles = E / 16;
  const long long* srcp = eidx;
  const long long* dstp = eidx + E;

  char* ws = (char*)d_ws;
  size_t off = 0;
  auto alloc = [&](size_t bytes) -> char* {
    char* p = ws + off;
    off = (off + bytes + 255) & ~(size_t)255;
    return p;
  };
  _Float16* Wk1h = (_Float16*)alloc(64 * 160 * 2);
  _Float16* Wv1h = (_Float16*)alloc(64 * 160 * 2);
  _Float16* Wk2h = (_Float16*)alloc(64 * 64 * 2);
  _Float16* Wv2h = (_Float16*)alloc(64 * 64 * 2);
  _Float16* Wq1h = (_Float16*)alloc(64 * 64 * 2);
  _Float16* Wq2h = (_Float16*)alloc(64 * 64 * 2);
  _Float16* WOh  = (_Float16*)alloc(64 * 64 * 2);
  _Float16* W1h  = (_Float16*)alloc(128 * 64 * 2);
  _Float16* W2h  = (_Float16*)alloc(64 * 128 * 2);
  float*    qbuf   = (float*)alloc((size_t)N * 64 * 4);
  float*    scores = (float*)alloc((size_t)E * 4 * 4);
  _Float16* vbuf   = (_Float16*)alloc((size_t)E * 64 * 2);
  unsigned* smax   = (unsigned*)alloc((size_t)N * 4 * 4);
  float*    denom  = (float*)alloc((size_t)N * 4 * 4);
  float*    agg    = (float*)alloc((size_t)N * 64 * 4);
  float*    xpre   = (float*)alloc((size_t)N * 64 * 4);
  float*    ypre   = (float*)alloc((size_t)N * 64 * 4);
  float*    bn     = (float*)alloc(512 * 4);
  // bn layout: sum1[0:64] sq1[64:128] sum2[128:192] sq2[192:256]
  //            mu1[256:320] rs1[320:384] mu2[384:448] rs2[448:512]
  float *sum1 = bn, *sq1 = bn + 64, *sum2 = bn + 128, *sq2 = bn + 192;
  float *mu1 = bn + 256, *rs1 = bn + 320, *mu2 = bn + 384, *rs2 = bn + 448;

  // 1) weight prep (fp32 row-major -> f16 column-major, K padded)
  auto prep = [&](const float* s, _Float16* d, int K, int Nn, int Kpad) {
    int tot = Nn * Kpad;
    prep_w_kernel<<<(tot + 255) / 256, 256, 0, stream>>>(s, d, K, Nn, Kpad);
  };
  prep(Wk1, Wk1h, 144, 64, 160);
  prep(Wv1, Wv1h, 144, 64, 160);
  prep(Wk2, Wk2h, 64, 64, 64);
  prep(Wv2, Wv2h, 64, 64, 64);
  prep(Wq1, Wq1h, 64, 64, 64);
  prep(Wq2, Wq2h, 64, 64, 64);
  prep(WO,  WOh,  64, 64, 64);
  prep(W1,  W1h,  64, 128, 64);
  prep(W2,  W2h,  128, 64, 128);

  // 2) zero-init scatter targets + BN accumulators
  hipMemsetAsync(smax, 0, (size_t)N * 4 * 4, stream);
  hipMemsetAsync(denom, 0, (size_t)N * 4 * 4, stream);
  hipMemsetAsync(agg, 0, (size_t)N * 64 * 4, stream);
  hipMemsetAsync(bn, 0, 256 * 4, stream);

  // 3) q MLP
  q_mlp_kernel<<<(Ntiles + WPB - 1) / WPB, TPB, 0, stream>>>(h, Wq1h, bq1, Wq2h, bq2, qbuf, Ntiles);
  // 4) edge K/V MLPs + scores + fused segment-max
  edge_kv_kernel<<<(Etiles + WPB - 1) / WPB, TPB, 0, stream>>>(
      h, ea, srcp, dstp, Wk1h, bk1, Wk2h, bk2, Wv1h, bv1, Wv2h, bv2,
      qbuf, scores, vbuf, smax, Etiles);
  // 5) fused exp / denominator / weighted-V scatter
  int E4 = E * 4;
  scatter_kernel<<<(E4 + 255) / 256, 256, 0, stream>>>(scores, dstp, smax, vbuf, denom, agg, E4);
  // 6) output projection + residual, BN1 stats
  node_proj_kernel<<<(Ntiles + WPB - 1) / WPB, TPB, 0, stream>>>(
      h, agg, denom, WOh, bO, xpre, sum1, sq1, Ntiles);
  bn_stats_kernel<<<1, 64, 0, stream>>>(sum1, sq1, mu1, rs1, 1.0f / (float)N);
  // 7) BN1 apply + FFN + residual, BN2 stats
  ffn_kernel<<<(Ntiles + WPB - 1) / WPB, TPB, 0, stream>>>(
      xpre, mu1, rs1, g1, be1, W1h, b1, W2h, b2, ypre, sum2, sq2, Ntiles);
  bn_stats_kernel<<<1, 64, 0, stream>>>(sum2, sq2, mu2, rs2, 1.0f / (float)N);
  // 8) BN2 apply -> output
  finalize_kernel<<<(N * 64 + 255) / 256, 256, 0, stream>>>(ypre, mu2, rs2, g2, be2, out, N * 64);

  (void)n_in; (void)out_size; (void)ws_size;
}
